// CODALayer_12249246728620
// MI455X (gfx1250) — matrix operand order repaired and checked by reading, not compiled, not run on hardware
//
// CODA layer for MI455X (gfx1250, wave32, WMMA). All heavy math is expressed as
// f16-in / f32-accumulate GEMMs on v_wmma_f32_16x16x32_f16:
//   * spectral convs = truncated-DFT matmul chains (batched complex GEMMs)
//   * attention      = real GEMMs (QK^T, P@V)
// Complex GEMM variants are template-specialized so the inner loop is
// branch-free: loads + v_wmma + unconditional WGP-scope prefetch only.
// Pointwise stages are VALU. Workspace use: ~378 MiB scratch from d_ws.

#include <hip/hip_runtime.h>
#include <hip/hip_bf16.h>
#include <math.h>

typedef _Float16 f16;
typedef _Float16 v16h __attribute__((ext_vector_type(16)));
typedef _Float16 v8h  __attribute__((ext_vector_type(8)));
typedef float    v8f  __attribute__((ext_vector_type(8)));

#define DEV __device__ __forceinline__

static constexpr int   NT  = 128;       // B*T tokens
static constexpr int   HW  = 128 * 128;
static constexpr float PI2 = 6.28318530717958647692f;

DEV float gelu_f(float v) { return 0.5f * v * (1.f + erff(v * 0.70710678118654752f)); }

DEV v8f wmma32(v16h a, v16h b, v8f c) {
  return __builtin_amdgcn_wmma_f32_16x16x32_f16(false, a, false, b, (short)0, c,
                                                false, false);
}

// A fragment (16x32, row-major, lda mult of 8): lane&15 = row, lane>>4 selects
// k-half. v[0..7] = A[row][kb+kh .. kb+kh+7], v[8..15] = A[row][kb+16+kh ..].
DEV v16h ld_a(const f16* __restrict__ base, int lda, int lane, int kb) {
  const f16* p = base + (size_t)(lane & 15) * lda + kb + ((lane >> 4) << 3);
  v8h lo = *(const v8h*)p;
  v8h hi = *(const v8h*)(p + 16);
  v16h r;
#pragma unroll
  for (int i = 0; i < 8; ++i) { r[i] = lo[i]; r[i + 8] = hi[i]; }
  return r;
}

// B fragment from B^T (N x K row-major): lane&15 = col, k order as in ld_a.
DEV v16h ld_bt(const f16* __restrict__ base, int ldb, int lane, int kb) {
  const f16* p = base + (size_t)(lane & 15) * ldb + kb + ((lane >> 4) << 3);
  v8h lo = *(const v8h*)p;
  v8h hi = *(const v8h*)(p + 16);
  v16h r;
#pragma unroll
  for (int i = 0; i < 8; ++i) { r[i] = lo[i]; r[i + 8] = hi[i]; }
  return r;
}

// ---------------- complex GEMM:  C = A @ B  (B given transposed) -------------
// One wave per 16x16 tile. Compile-time variants: HAS_BI (B complex vs real),
// F16OUT (f16 vs f32 C), WRITE_CI (full complex vs real-part-only output).
// A is always complex here (DFT matrices / complex intermediates).
template <bool HAS_BI, bool F16OUT, bool WRITE_CI>
__global__ __launch_bounds__(32)
void k_cgemm(const f16* __restrict__ Ar, const f16* __restrict__ Ai,
             long long sA, int lda,
             const f16* __restrict__ Br, const f16* __restrict__ Bi,
             long long sB, int ldb,
             void* __restrict__ Crp, void* __restrict__ Cip,
             long long sC, int ldc, int K) {
  const int lane = threadIdx.x;
  const long long bz = blockIdx.z;
  const f16* arp = Ar + bz * sA + (size_t)(blockIdx.y * 16) * lda;
  const f16* aip = Ai + bz * sA + (size_t)(blockIdx.y * 16) * lda;
  const f16* brp = Br + bz * sB + (size_t)(blockIdx.x * 16) * ldb;
  const f16* bip = HAS_BI ? Bi + bz * sB + (size_t)(blockIdx.x * 16) * ldb : nullptr;
  v8f rr = {}, ri = {}, ir = {}, ii = {};
#pragma unroll 2
  for (int kb = 0; kb < K; kb += 32) {
    // Speculative near-cache prefetch of the next k-block (no guard: prefetch
    // past the end is architecturally harmless and avoids an in-loop branch).
    __builtin_prefetch(arp + (size_t)(lane & 15) * lda + kb + 32, 0, 3);
    __builtin_prefetch(brp + (size_t)(lane & 15) * ldb + kb + 32, 0, 3);
    v16h af  = ld_a(arp, lda, lane, kb);
    v16h aif = ld_a(aip, lda, lane, kb);
    v16h bf  = ld_bt(brp, ldb, lane, kb);
    rr = wmma32(af, bf, rr);
    ir = wmma32(aif, bf, ir);
    if (HAS_BI) {
      __builtin_prefetch(bip + (size_t)(lane & 15) * ldb + kb + 32, 0, 3);
      v16h bif = ld_bt(bip, ldb, lane, kb);
      ri = wmma32(af, bif, ri);
      ii = wmma32(aif, bif, ii);
    }
  }
  const int row0 = blockIdx.y * 16 + ((lane >> 4) << 3);
  const int col  = blockIdx.x * 16 + (lane & 15);
#pragma unroll
  for (int r = 0; r < 8; ++r) {
    float cr = HAS_BI ? (rr[r] - ii[r]) : rr[r];
    float ci = HAS_BI ? (ri[r] + ir[r]) : ir[r];
    size_t off = (size_t)bz * sC + (size_t)(row0 + r) * ldc + col;
    if (F16OUT) {
      ((f16*)Crp)[off] = (f16)cr;
      if (WRITE_CI) ((f16*)Cip)[off] = (f16)ci;
    } else {
      ((float*)Crp)[off] = cr;
      if (WRITE_CI) ((float*)Cip)[off] = ci;
    }
  }
}

// ---------------- real GEMM (attention) --------------------------------------
__global__ __launch_bounds__(32)
void k_gemm(const f16* __restrict__ A, long long sA, int lda,
            const f16* __restrict__ B, long long sB, int ldb,
            float* __restrict__ C, long long sC, int ldc, int K) {
  const int lane = threadIdx.x;
  const long long bz = blockIdx.z;
  const f16* ap = A + bz * sA + (size_t)(blockIdx.y * 16) * lda;
  const f16* bp = B + bz * sB + (size_t)(blockIdx.x * 16) * ldb;
  v8f acc = {};
#pragma unroll 2
  for (int kb = 0; kb < K; kb += 32) {
    __builtin_prefetch(ap + (size_t)(lane & 15) * lda + kb + 32, 0, 3);
    __builtin_prefetch(bp + (size_t)(lane & 15) * ldb + kb + 32, 0, 3);
    acc = wmma32(ld_a(ap, lda, lane, kb), ld_bt(bp, ldb, lane, kb), acc);
  }
  const int row0 = blockIdx.y * 16 + ((lane >> 4) << 3);
  const int col  = blockIdx.x * 16 + (lane & 15);
#pragma unroll
  for (int r = 0; r < 8; ++r)
    C[(size_t)bz * sC + (size_t)(row0 + r) * ldc + col] = acc[r];
}

// ---------------- instance norm (per image), optional post-norm residual -----
__global__ void k_inorm(const float* __restrict__ x, const float* __restrict__ g,
                        const float* __restrict__ b, int C, int HW_,
                        const float* __restrict__ res, float* __restrict__ out,
                        f16* __restrict__ outT, int Hd, int Wd) {
  __shared__ float sm[16];
  const int n = blockIdx.x;
  const int c = n % C;
  const float* xp = x + (size_t)n * HW_;
  float s = 0.f, s2 = 0.f;
  for (int i = threadIdx.x; i < HW_; i += blockDim.x) {
    float v = xp[i]; s += v; s2 += v * v;
  }
  for (int m = 16; m; m >>= 1) { s += __shfl_xor(s, m, 32); s2 += __shfl_xor(s2, m, 32); }
  const int wid = threadIdx.x >> 5, nw = blockDim.x >> 5;
  if ((threadIdx.x & 31) == 0) { sm[wid] = s; sm[8 + wid] = s2; }
  __syncthreads();
  if (threadIdx.x == 0) {
    float a = 0.f, a2 = 0.f;
    for (int i = 0; i < nw; ++i) { a += sm[i]; a2 += sm[8 + i]; }
    sm[0] = a; sm[8] = a2;
  }
  __syncthreads();
  const float mu  = sm[0] / HW_;
  const float var = sm[8] / HW_ - mu * mu;
  const float rs  = rsqrtf(var + 1e-5f) * g[c];
  const float bb  = b[c];
  for (int i = threadIdx.x; i < HW_; i += blockDim.x) {
    float v = (xp[i] - mu) * rs + bb;
    if (res) v += res[(size_t)n * HW_ + i];
    if (out) out[(size_t)n * HW_ + i] = v;
    if (outT) {
      int hh = i / Wd, ww = i % Wd;
      outT[(size_t)n * HW_ + (size_t)ww * Hd + hh] = (f16)v;
    }
  }
}

// ---------------- 1x1 conv over channels, optional add + gelu + f16-T copy ---
__global__ void k_conv1x1(const float* __restrict__ x, const float* __restrict__ w,
                          const float* __restrict__ bias, int cin, int cout,
                          int HW_, const float* __restrict__ add, int act,
                          float* __restrict__ out, f16* __restrict__ outT,
                          int Hd, int Wd, int Nimg) {
  size_t idx = (size_t)blockIdx.x * blockDim.x + threadIdx.x;
  size_t tot = (size_t)Nimg * cout * HW_;
  if (idx >= tot) return;
  int p = (int)(idx % HW_);
  size_t t = idx / HW_;
  int o = (int)(t % cout);
  size_t nimg = t / cout;
  float v = bias ? bias[o] : 0.f;
  for (int i = 0; i < cin; ++i)
    v += x[((size_t)nimg * cin + i) * HW_ + p] * w[o * cin + i];
  if (add) v += add[idx];
  if (act) v = gelu_f(v);
  if (out) out[idx] = v;
  if (outT) {
    int hh = p / Wd, ww = p % Wd;
    outT[((size_t)nimg * cout + o) * HW_ + (size_t)ww * Hd + hh] = (f16)v;
  }
}

__global__ void k_addact(const float* __restrict__ a, const float* __restrict__ b,
                         int act, size_t n, float* __restrict__ out,
                         f16* __restrict__ outT, int Hd, int Wd) {
  size_t idx = (size_t)blockIdx.x * blockDim.x + threadIdx.x;
  if (idx >= n) return;
  float v = a[idx] + (b ? b[idx] : 0.f);
  if (act) v = gelu_f(v);
  if (out) out[idx] = v;
  if (outT) {
    size_t HW_ = (size_t)Hd * Wd;
    size_t nimg = idx / HW_;
    int p = (int)(idx % HW_);
    int hh = p / Wd, ww = p % Wd;
    outT[nimg * HW_ + (size_t)ww * Hd + hh] = (f16)v;
  }
}

__global__ void k_pool2(const float* __restrict__ x, float* __restrict__ out,
                        int S, size_t Nc) {
  size_t idx = (size_t)blockIdx.x * blockDim.x + threadIdx.x;
  size_t tot = Nc * (size_t)S * S;
  if (idx >= tot) return;
  int xx = (int)(idx % S);
  int yy = (int)((idx / S) % S);
  size_t n = idx / ((size_t)S * S);
  const float* p = x + n * (size_t)(4 * S * S) + (size_t)(2 * yy) * (2 * S) + 2 * xx;
  out[idx] = 0.25f * (p[0] + p[1] + p[2 * S] + p[2 * S + 1]);
}

// ---------------- spectral mode mixing (complex channel einsum) --------------
// Xf: (Ntok*cin) batches of (2m x 32) complex f32. ZT: (Ntok*cout) batches of
// (32 x 32) complex f16, TRANSPOSED (row = col-freq q, col = row-freq p), padded
// with zeros so downstream GEMMs can use K=N=32.
__global__ void k_mode_mix(const float* __restrict__ Xr, const float* __restrict__ Xi,
                           const float* __restrict__ w1, const float* __restrict__ w2,
                           int cin, int cout, int m, int Ntok,
                           f16* __restrict__ Zr, f16* __restrict__ Zi) {
  size_t idx = (size_t)blockIdx.x * blockDim.x + threadIdx.x;
  size_t tot = (size_t)Ntok * cout * 1024;
  if (idx >= tot) return;
  int p = (int)(idx & 31);
  int q = (int)((idx >> 5) & 31);
  size_t t2 = idx >> 10;
  int o = (int)(t2 % cout);
  size_t n = t2 / cout;
  float zr = 0.f, zi = 0.f;
  if (p < 2 * m && q < m) {
    const int pm = (p < m) ? p : p - m;
    const float* wt = (p < m) ? w1 : w2;
    const int xbs = 2 * m * 32;
    for (int i = 0; i < cin; ++i) {
      size_t xo = (n * cin + i) * (size_t)xbs + (size_t)p * 32 + q;
      float xr = Xr[xo], xi = Xi[xo];
      size_t wo = ((((size_t)i * cout + o) * m + pm) * m + q) * 2;
      float wr = wt[wo], wi = wt[wo + 1];
      zr += xr * wr - xi * wi;
      zi += xr * wi + xi * wr;
    }
  }
  size_t zo = (n * cout + o) * 1024 + (size_t)q * 32 + p;
  Zr[zo] = (f16)zr;
  Zi[zo] = (f16)zi;
}

// ---------------- head flatten / unflatten / softmax -------------------------
__global__ void k_flat_qk(const float* __restrict__ img, f16* __restrict__ out, int A_) {
  size_t idx = (size_t)blockIdx.x * blockDim.x + threadIdx.x;
  size_t tot = (size_t)16 * 32 * A_;
  if (idx >= tot) return;
  int e = (int)(idx % A_);
  size_t t1 = idx / A_;
  int t = (int)(t1 % 32);
  size_t t2 = t1 / 32;
  int h = (int)(t2 % 4), b = (int)(t2 / 4);
  out[idx] = (f16)img[(((size_t)(b * 32 + t)) * 4 + h) * A_ + e];
}

__global__ void k_flat_vT(const float* __restrict__ img, f16* __restrict__ out, int A_) {
  size_t idx = (size_t)blockIdx.x * blockDim.x + threadIdx.x;
  size_t tot = (size_t)16 * 32 * A_;
  if (idx >= tot) return;
  int t = (int)(idx % 32);
  size_t t1 = idx / 32;
  int e = (int)(t1 % A_);
  size_t bh = t1 / A_;
  int h = (int)(bh % 4), b = (int)(bh / 4);
  out[idx] = (f16)img[(((size_t)(b * 32 + t)) * 4 + h) * (size_t)A_ + e];
}

__global__ void k_unflat(const float* __restrict__ o, float* __restrict__ img,
                         f16* __restrict__ imgT) {
  size_t idx = (size_t)blockIdx.x * blockDim.x + threadIdx.x;
  size_t tot = (size_t)16 * 32 * HW;
  if (idx >= tot) return;
  int e = (int)(idx % HW);
  size_t t1 = idx / HW;
  int t = (int)(t1 % 32);
  int h = (int)((t1 / 32) % 4);
  int b = (int)(t1 / 128);
  float v = o[idx];
  size_t im = ((size_t)(b * 32 + t)) * 4 + h;
  img[im * HW + e] = v;
  int hh = e >> 7, ww = e & 127;
  imgT[im * HW + (size_t)ww * 128 + hh] = (f16)v;
}

__global__ __launch_bounds__(32)
void k_softmax32(const float* __restrict__ s, f16* __restrict__ p, float scale) {
  int row = blockIdx.x, lane = threadIdx.x;
  float v = s[(size_t)row * 32 + lane] * scale;
  float m = v;
  for (int d = 16; d; d >>= 1) m = fmaxf(m, __shfl_xor(m, d, 32));
  float e = __expf(v - m);
  float sum = e;
  for (int d = 16; d; d >>= 1) sum += __shfl_xor(sum, d, 32);
  p[(size_t)row * 32 + lane] = (f16)(e / sum);
}

// ---------------- DFT table builders (f16) -----------------------------------
__global__ void k_tab_fr(f16* r, f16* i, int m1, int Hdim, float scale) {
  int idx = blockIdx.x * blockDim.x + threadIdx.x;
  int tot = 2 * m1 * Hdim;
  if (idx >= tot) return;
  int hc = idx % Hdim, row = idx / Hdim;
  int f = row < m1 ? row : Hdim - 2 * m1 + row;
  float th = PI2 * (float)f * (float)hc / (float)Hdim;
  r[idx] = (f16)(scale * cosf(th));
  i[idx] = (f16)(-scale * sinf(th));
}
__global__ void k_tab_fcT(f16* r, f16* i, int m2, int Wdim) {
  int idx = blockIdx.x * blockDim.x + threadIdx.x;
  int tot = 32 * Wdim;
  if (idx >= tot) return;
  int w = idx % Wdim, q = idx / Wdim;
  if (q < m2) {
    float th = PI2 * (float)q * (float)w / (float)Wdim;
    r[idx] = (f16)cosf(th);
    i[idx] = (f16)(-sinf(th));
  } else { r[idx] = (f16)0.f; i[idx] = (f16)0.f; }
}
__global__ void k_tab_gr(f16* r, f16* i, int oh, int m1) {
  int idx = blockIdx.x * blockDim.x + threadIdx.x;
  int tot = oh * 32;
  if (idx >= tot) return;
  int p = idx % 32, hh = idx / 32;
  if (p < 2 * m1) {
    int f = p < m1 ? p : oh - 2 * m1 + p;
    float th = PI2 * (float)f * (float)hh / (float)oh;
    r[idx] = (f16)cosf(th);
    i[idx] = (f16)sinf(th);
  } else { r[idx] = (f16)0.f; i[idx] = (f16)0.f; }
}
__global__ void k_tab_gcT(f16* r, f16* i, int ow, int m2) {
  int idx = blockIdx.x * blockDim.x + threadIdx.x;
  int tot = ow * 32;
  if (idx >= tot) return;
  int q = idx % 32, w = idx / 32;
  if (q < m2) {
    float wq = (q == 0 || 2 * q == ow) ? 1.f : 2.f;
    float th = PI2 * (float)q * (float)w / (float)ow;
    r[idx] = (f16)(wq * cosf(th));
    i[idx] = (f16)(wq * sinf(th));
  } else { r[idx] = (f16)0.f; i[idx] = (f16)0.f; }
}

// =============================== host side ===================================
struct FnoP {
  const float *w1, *w2, *skip_w, *skip_b, *mlp_w, *mlp_b, *fc1_w, *fc1_b, *fc2_w, *fc2_b;
};
struct SpecCfg {
  const f16 *Frr, *Fri;    // (2m x 128) forward row DFT, lda=128 (scale folded)
  const f16 *FcTr, *FcTi;  // (32 x 128) forward col DFT^T, zero-padded rows
  const f16 *Grr, *Gri;    // (oh x 32) inverse row DFT, zero-padded cols
  const f16 *GcTr, *GcTi;  // (ow x 32) inverse col DFT^T (with rfft weights)
  int m, oh, ow;
};
struct Scratch {
  f16 *T1r, *T1i, *ZTr, *ZTi, *Y1r, *Y1i;
  float *Xfr, *Xfi;
  float *yspec, *skip, *tmp, *ms;
};

static inline dim3 nb(size_t t) { return dim3((unsigned)((t + 255) / 256)); }

static void run_spectral(hipStream_t st, const SpecCfg& c, const f16* xT,
                         int cin, int cout, const float* w1, const float* w2,
                         const Scratch& s, float* yout) {
  const int twom = 2 * c.m;
  dim3 wv(32);
  // S1: T1 = Fr @ x   (x real, fed transposed) -> complex f16
  k_cgemm<false, true, true><<<dim3(8, twom / 16, NT * cin), wv, 0, st>>>(
      c.Frr, c.Fri, 0, 128, xT, nullptr, (long long)HW, 128,
      s.T1r, s.T1i, (long long)twom * 128, 128, 128);
  // S2: Xf = T1 @ Fc  -> complex f32
  k_cgemm<true, false, true><<<dim3(2, twom / 16, NT * cin), wv, 0, st>>>(
      s.T1r, s.T1i, (long long)twom * 128, 128, c.FcTr, c.FcTi, 0, 128,
      s.Xfr, s.Xfi, (long long)twom * 32, 32, 128);
  // S3: per-mode complex channel mixing (tiny cin/cout) -> ZT (padded, f16)
  k_mode_mix<<<nb((size_t)NT * cout * 1024), 256, 0, st>>>(
      s.Xfr, s.Xfi, w1, w2, cin, cout, c.m, NT, s.ZTr, s.ZTi);
  // S4: Y1 = Gr @ Z   -> complex f16
  k_cgemm<true, true, true><<<dim3(2, c.oh / 16, NT * cout), wv, 0, st>>>(
      c.Grr, c.Gri, 0, 32, s.ZTr, s.ZTi, 1024, 32,
      s.Y1r, s.Y1i, (long long)c.oh * 32, 32, 32);
  // S5: y = Re(Y1 @ Gc) -> f32
  k_cgemm<true, false, false><<<dim3(c.ow / 16, c.oh / 16, NT * cout), wv, 0, st>>>(
      s.Y1r, s.Y1i, (long long)c.oh * 32, 32, c.GcTr, c.GcTi, 0, 32,
      yout, nullptr, (long long)c.oh * c.ow, c.ow, 32);
}

static void run_fno(hipStream_t st, const SpecCfg& c, const FnoP& p,
                    const float* x, const f16* xT, int cin, int cout, int hidden,
                    int act, int pool, const Scratch& s,
                    const float* n0g, const float* n0b,
                    const float* n1g, const float* n1b, float* out) {
  const int HWin = HW;
  const int HWout = c.oh * c.ow;
  const size_t totout = (size_t)NT * cout * HWout;
  if (pool) {  // skip paths downsampled to out grid (approximates fourier_resample)
    k_conv1x1<<<nb((size_t)NT * cout * HWin), 256, 0, st>>>(
        x, p.skip_w, p.skip_b, cin, cout, HWin, nullptr, 0, s.tmp, nullptr, 0, 0, NT);
    k_pool2<<<nb(totout), 256, 0, st>>>(s.tmp, s.skip, c.ow, (size_t)NT * cout);
    k_conv1x1<<<nb((size_t)NT * cout * HWin), 256, 0, st>>>(
        x, p.mlp_w, p.mlp_b, cin, cout, HWin, nullptr, 0, s.tmp, nullptr, 0, 0, NT);
    k_pool2<<<nb(totout), 256, 0, st>>>(s.tmp, s.ms, c.ow, (size_t)NT * cout);
  } else {
    k_conv1x1<<<nb(totout), 256, 0, st>>>(
        x, p.skip_w, p.skip_b, cin, cout, HWout, nullptr, 0, s.skip, nullptr, 0, 0, NT);
    k_conv1x1<<<nb(totout), 256, 0, st>>>(
        x, p.mlp_w, p.mlp_b, cin, cout, HWout, nullptr, 0, s.ms, nullptr, 0, 0, NT);
  }
  run_spectral(st, c, xT, cin, cout, p.w1, p.w2, s, s.yspec);
  if (n0g)
    k_inorm<<<NT * cout, 256, 0, st>>>(s.yspec, n0g, n0b, cout, HWout, nullptr,
                                       s.yspec, nullptr, 0, 0);
  float* ybuf = s.tmp;   // tmp free by now
  k_addact<<<nb(totout), 256, 0, st>>>(s.yspec, s.skip, act, totout, ybuf,
                                       nullptr, 0, 0);
  float* hdn = s.skip;   // skip free by now
  k_conv1x1<<<nb((size_t)NT * hidden * HWout), 256, 0, st>>>(
      ybuf, p.fc1_w, p.fc1_b, cout, hidden, HWout, nullptr, act, hdn, nullptr, 0, 0, NT);
  k_conv1x1<<<nb(totout), 256, 0, st>>>(
      hdn, p.fc2_w, p.fc2_b, hidden, cout, HWout, s.ms, 0, out, nullptr, 0, 0, NT);
  if (n1g)
    k_inorm<<<NT * cout, 256, 0, st>>>(out, n1g, n1b, cout, HWout, nullptr, out,
                                       nullptr, 0, 0);
}

extern "C" void kernel_launch(void* const* d_in, const int* in_sizes, int n_in,
                              void* d_out, int out_size, void* d_ws, size_t ws_size,
                              hipStream_t stream) {
  (void)in_sizes; (void)n_in; (void)out_size; (void)ws_size;
  const float* x = (const float*)d_in[0];
  auto P = [&](int i) { return (const float*)d_in[i]; };
  auto FP = [&](int b) {
    FnoP p = {P(b), P(b + 1), P(b + 2), P(b + 3), P(b + 4),
              P(b + 5), P(b + 6), P(b + 7), P(b + 8), P(b + 9)};
    return p;
  };
  FnoP pK = FP(1), pQ = FP(11), pV = FP(21), pProj = FP(31), pM0 = FP(41), pM1 = FP(51);
  const float *n1g = P(61), *n1b = P(62);       // norm1
  const float *ang = P(63), *anb = P(64);       // attn_norm
  const float *mig = P(65), *mib = P(66);       // mix_in
  const float *mog = P(67), *mob = P(68);       // mix_out
  const float *m0ag = P(69), *m0ab = P(70), *m0bg = P(71), *m0bb = P(72);
  const float *m1ag = P(73), *m1ab = P(74), *m1bg = P(75), *m1bb = P(76);

  char* wp = (char*)d_ws;
  size_t used = 0;
  auto alloc = [&](size_t bytes) -> void* {
    used = (used + 255) & ~(size_t)255;
    void* r = wp + used;
    used += bytes;
    return r;
  };
  auto aF = [&](size_t n) { return (float*)alloc(n * 4); };
  auto aH = [&](size_t n) { return (f16*)alloc(n * 2); };

  // ---- DFT tables -----------------------------------------------------------
  const float sc = 1.f / (float)HW;  // rfft2 norm='forward'
  f16 *FrKr = aH(16 * 128), *FrKi = aH(16 * 128);
  f16 *FrPr = aH(32 * 128), *FrPi = aH(32 * 128);
  f16 *FcKr = aH(32 * 128), *FcKi = aH(32 * 128);
  f16 *FcPr = aH(32 * 128), *FcPi = aH(32 * 128);
  f16 *GrKr = aH(64 * 32),  *GrKi = aH(64 * 32);
  f16 *GrVr = aH(128 * 32), *GrVi = aH(128 * 32);
  f16 *GrPr = aH(128 * 32), *GrPi = aH(128 * 32);
  f16 *GcKr = aH(64 * 32),  *GcKi = aH(64 * 32);
  f16 *GcVr = aH(128 * 32), *GcVi = aH(128 * 32);
  f16 *GcPr = aH(128 * 32), *GcPi = aH(128 * 32);
  k_tab_fr<<<nb(16 * 128), 256, 0, stream>>>(FrKr, FrKi, 8, 128, sc);
  k_tab_fr<<<nb(32 * 128), 256, 0, stream>>>(FrPr, FrPi, 16, 128, sc);
  k_tab_fcT<<<nb(32 * 128), 256, 0, stream>>>(FcKr, FcKi, 8, 128);
  k_tab_fcT<<<nb(32 * 128), 256, 0, stream>>>(FcPr, FcPi, 16, 128);
  k_tab_gr<<<nb(64 * 32), 256, 0, stream>>>(GrKr, GrKi, 64, 8);
  k_tab_gr<<<nb(128 * 32), 256, 0, stream>>>(GrVr, GrVi, 128, 8);
  k_tab_gr<<<nb(128 * 32), 256, 0, stream>>>(GrPr, GrPi, 128, 16);
  k_tab_gcT<<<nb(64 * 32), 256, 0, stream>>>(GcKr, GcKi, 64, 8);
  k_tab_gcT<<<nb(128 * 32), 256, 0, stream>>>(GcVr, GcVi, 128, 8);
  k_tab_gcT<<<nb(128 * 32), 256, 0, stream>>>(GcPr, GcPi, 128, 16);

  SpecCfg cK = {FrKr, FrKi, FcKr, FcKi, GrKr, GrKi, GcKr, GcKi, 8, 64, 64};
  SpecCfg cV = {FrKr, FrKi, FcKr, FcKi, GrVr, GrVi, GcVr, GcVi, 8, 128, 128};
  SpecCfg cP = {FrPr, FrPi, FcPr, FcPi, GrPr, GrPi, GcPr, GcPi, 16, 128, 128};

  // ---- scratch --------------------------------------------------------------
  Scratch s;
  s.T1r = aH((size_t)NT * 4 * 32 * 128); s.T1i = aH((size_t)NT * 4 * 32 * 128);
  s.Xfr = aF((size_t)NT * 4 * 32 * 32);  s.Xfi = aF((size_t)NT * 4 * 32 * 32);
  s.ZTr = aH((size_t)NT * 4 * 1024);     s.ZTi = aH((size_t)NT * 4 * 1024);
  s.Y1r = aH((size_t)NT * 4 * 128 * 32); s.Y1i = aH((size_t)NT * 4 * 128 * 32);
  s.yspec = aF((size_t)NT * 4 * HW);
  s.skip  = aF((size_t)NT * 4 * HW);
  s.tmp   = aF((size_t)NT * 4 * HW);
  s.ms    = aF((size_t)NT * 4 * HW);

  float* tn   = aF((size_t)NT * HW);
  f16*   tnT  = aH((size_t)NT * HW);
  float* kimg = aF((size_t)NT * 4 * 4096);
  float* qimg = aF((size_t)NT * 4 * 4096);
  float* vimg = aF((size_t)NT * 4 * HW);
  f16*   kf   = aH((size_t)16 * 32 * 4096);
  f16*   qf   = aH((size_t)16 * 32 * 4096);
  f16*   vT   = aH((size_t)16 * 32 * HW);
  float* scr  = aF((size_t)16 * 32 * 32);
  f16*   prb  = aH((size_t)16 * 32 * 32);
  float* obuf = aF((size_t)16 * 32 * HW);
  float* oimg = aF((size_t)NT * 4 * HW);
  f16*   oT   = aH((size_t)NT * 4 * HW);
  float* pout = aF((size_t)NT * HW);
  float* apre = aF((size_t)NT * HW);
  float* attn = aF((size_t)NT * HW);
  float* ymix = aF((size_t)NT * HW);
  f16*   ymixT = aH((size_t)NT * HW);
  float* m0o  = aF((size_t)NT * HW);
  float* m0g_ = aF((size_t)NT * HW);
  f16*   m0gT = aH((size_t)NT * HW);
  float* m1o  = aF((size_t)NT * HW);

  // ---- pipeline -------------------------------------------------------------
  // tn = inorm(tokens)
  k_inorm<<<NT, 256, 0, stream>>>(x, n1g, n1b, 1, HW, nullptr, tn, tnT, 128, 128);
  // K / Q (out 64x64), V (out 128x128); act = identity
  run_fno(stream, cK, pK, tn, tnT, 1, 4, 4, 0, 1, s, nullptr, nullptr, nullptr, nullptr, kimg);
  run_fno(stream, cK, pQ, tn, tnT, 1, 4, 4, 0, 1, s, nullptr, nullptr, nullptr, nullptr, qimg);
  run_fno(stream, cV, pV, tn, tnT, 1, 4, 4, 0, 0, s, nullptr, nullptr, nullptr, nullptr, vimg);

  // attention
  k_flat_qk<<<nb((size_t)16 * 32 * 4096), 256, 0, stream>>>(kimg, kf, 4096);
  k_flat_qk<<<nb((size_t)16 * 32 * 4096), 256, 0, stream>>>(qimg, qf, 4096);
  k_flat_vT<<<nb((size_t)16 * 32 * HW), 256, 0, stream>>>(vimg, vT, HW);
  k_gemm<<<dim3(2, 2, 16), 32, 0, stream>>>(qf, (long long)32 * 4096, 4096,
                                            kf, (long long)32 * 4096, 4096,
                                            scr, 1024, 32, 4096);
  k_softmax32<<<16 * 32, 32, 0, stream>>>(scr, prb, 1.f / 64.f);
  k_gemm<<<dim3(1024, 2, 16), 32, 0, stream>>>(prb, 1024, 32,
                                               vT, (long long)HW * 32, 32,
                                               obuf, (long long)32 * HW, HW, 32);
  k_unflat<<<nb((size_t)16 * 32 * HW), 256, 0, stream>>>(obuf, oimg, oT);

  // multi_head_proj, residual, norms
  run_fno(stream, cP, pProj, oimg, oT, 4, 1, 1, 0, 0, s, nullptr, nullptr, nullptr, nullptr, pout);
  k_addact<<<nb((size_t)NT * HW), 256, 0, stream>>>(pout, x, 0, (size_t)NT * HW,
                                                    apre, nullptr, 0, 0);
  k_inorm<<<NT, 256, 0, stream>>>(apre, ang, anb, 1, HW, nullptr, attn, nullptr, 0, 0);
  k_inorm<<<NT, 256, 0, stream>>>(attn, mig, mib, 1, HW, nullptr, ymix, ymixT, 128, 128);

  // 2-layer mixer (gelu, inner norms)
  run_fno(stream, cP, pM0, ymix, ymixT, 1, 1, 1, 1, 0, s, m0ag, m0ab, m0bg, m0bb, m0o);
  k_addact<<<nb((size_t)NT * HW), 256, 0, stream>>>(m0o, nullptr, 1, (size_t)NT * HW,
                                                    m0g_, m0gT, 128, 128);
  run_fno(stream, cP, pM1, m0g_, m0gT, 1, 1, 1, 1, 0, s, m1ag, m1ab, m1bg, m1bb, m1o);

  // out = inorm(y, mix_out) + attn
  k_inorm<<<NT, 256, 0, stream>>>(m1o, mog, mob, 1, HW, attn, (float*)d_out,
                                  nullptr, 0, 0);
}
